// ConditionalVAE_73581379715145
// MI455X (gfx1250) — matrix operand order repaired and checked by reading, not compiled, not run on hardware
//
#include <hip/hip_runtime.h>

// ---------------- constants (from reference) ----------------
#define NS   256      // num_samples (batch)
#define ML   128      // max_len
#define VOCAB 128
#define EMB  256
#define PE   1280
#define HID  1024
#define LAT  128
#define BOS  1

typedef __bf16 bf16_t;
typedef __attribute__((ext_vector_type(16))) __bf16 v16bf;
typedef __attribute__((ext_vector_type(8)))  __bf16 v8bf;
typedef __attribute__((ext_vector_type(8)))  float  v8f;

__device__ __forceinline__ bf16_t f2bf(float f) {
    unsigned u = __builtin_bit_cast(unsigned, f);
    unsigned r = u + 0x7fffu + ((u >> 16) & 1u);   // round-to-nearest-even
    unsigned short h = (unsigned short)(r >> 16);
    return __builtin_bit_cast(bf16_t, h);
}

// ---------------- f32 -> bf16 row-major convert ----------------
__global__ __launch_bounds__(256) void cvt_bf16_kernel(const float* __restrict__ src,
                                                       bf16_t* __restrict__ dst, int n) {
    int i = blockIdx.x * 256 + threadIdx.x;
    if (i < n) dst[i] = f2bf(src[i]);
}

// ---------------- swizzle a [K,N] (or transposed [N,K]) f32 weight into
// WMMA B-fragment order: per 32x16 (KxN) tile, lane-major, 16 bf16/lane -----
__global__ __launch_bounds__(256) void swizzleB_kernel(const float* __restrict__ src,
                                                       bf16_t* __restrict__ dst,
                                                       int K, int N, int trans) {
    int i = blockIdx.x * 256 + threadIdx.x;
    if (i >= K * N) return;
    int k = i / N, n = i % N;
    int nt = n >> 4, col = n & 15;
    int kt = k >> 5, kk = k & 31;
    int lane = col + ((kk & 16) ? 16 : 0);
    int j = kk & 15;
    int KT = K >> 5;
    size_t di = (((size_t)nt * KT + kt) * 32 + lane) * 16 + j;
    float v = trans ? src[(size_t)n * K + k] : src[(size_t)k * N + n];
    dst[di] = f2bf(v);
}

// ---------------- matvec: out[j] = sum_k vec[k]*W[k,j] + b[j] ----------------
__global__ __launch_bounds__(256) void matvec_kernel(const float* __restrict__ vec,
                                                     const float* __restrict__ W,
                                                     const float* __restrict__ b,
                                                     float* __restrict__ out, int K, int N) {
    int j = blockIdx.x * 256 + threadIdx.x;
    if (j >= N) return;
    float s = b ? b[j] : 0.f;
    for (int k = 0; k < K; ++k) s += vec[k] * W[(size_t)k * N + j];
    out[j] = s;
}

// ---------------- core bf16 WMMA GEMM: LDS-shared B, double-buffered -------
// 4 waves / block; block computes 64(M) x 64(N). Per K-step each thread
// stages its lane-fragment of ONE B n-tile into LDS; all 4 waves consume all
// 4 tiles (4x less L2 traffic on weights). Double-buffered LDS: one barrier
// per K-step, next step's global loads issued before this step's WMMAs.
// Optional row-gather fuses the embedding lookup into the layer-0 GEMM.
__global__ __launch_bounds__(128) void wmma_gemm_lds_kernel(
        const bf16_t* __restrict__ A, const bf16_t* __restrict__ Bswz,
        const float* __restrict__ bias, float* __restrict__ C,
        bf16_t* __restrict__ Cbf,
        const int* __restrict__ gather, const bf16_t* __restrict__ gtable,
        int M, int N, int K, int relu) {
    __shared__ v8bf Bsh[2][4 * 32 * 2];       // two 4 KB buffers

    const int tid  = threadIdx.x;
    const int wave = tid >> 5;
    const int lane = tid & 31;
    const int mt   = blockIdx.x * 4 + wave;   // this wave's 16-row tile
    const int ntg  = blockIdx.y;              // 64-col group
    const int KT   = K >> 5;
    const int arow = lane & 15;
    const int koff = (lane >> 4) * 16;
    const int grow = mt * 16 + arow;
    const bf16_t* Ap = gather ? (gtable + (size_t)gather[grow] * K)
                              : (A + (size_t)grow * K);
    const int nt_mine = ntg * 4 + wave;       // the n-tile this thread stages
    const bf16_t* Bbase = Bswz + (((size_t)nt_mine * KT) * 32 + lane) * 16;
    const int myidx = (wave * 32 + lane) * 2;

    v8f acc[4] = {v8f{}, v8f{}, v8f{}, v8f{}};

    // ---- prologue: fetch K-step 0 fragments, stage B into buffer 0 ----
    v8bf bn0 = *(const v8bf*)(Bbase);
    v8bf bn1 = *(const v8bf*)(Bbase + 8);
    v8bf an0 = *(const v8bf*)(Ap + koff);
    v8bf an1 = *(const v8bf*)(Ap + koff + 8);
    Bsh[0][myidx + 0] = bn0;
    Bsh[0][myidx + 1] = bn1;
    int cur = 0;

    for (int kt = 0; kt < KT; ++kt) {
        v8bf a0 = an0, a1 = an1;
        __syncthreads();                      // publish Bsh[cur]; prev readers done
        if (kt + 1 < KT) {                    // issue next step's global loads now
            const bf16_t* Bp = Bbase + (size_t)(kt + 1) * 32 * 16;
            bn0 = *(const v8bf*)(Bp);
            bn1 = *(const v8bf*)(Bp + 8);
            const bf16_t* Apn = Ap + (kt + 1) * 32 + koff;
            an0 = *(const v8bf*)(Apn);
            an1 = *(const v8bf*)(Apn + 8);
        }
        v16bf a = __builtin_shufflevector(a0, a1,
                    0,1,2,3,4,5,6,7,8,9,10,11,12,13,14,15);
        // read ALL four B tiles first -> one DS wait amortized over 4 WMMAs
        v16bf bt[4];
#pragma unroll
        for (int t = 0; t < 4; ++t) {
            v8bf blo = Bsh[cur][(t * 32 + lane) * 2 + 0];
            v8bf bhi = Bsh[cur][(t * 32 + lane) * 2 + 1];
            bt[t] = __builtin_shufflevector(blo, bhi,
                        0,1,2,3,4,5,6,7,8,9,10,11,12,13,14,15);
        }
#pragma unroll
        for (int t = 0; t < 4; ++t)
            acc[t] = __builtin_amdgcn_wmma_f32_16x16x32_bf16(
                         false, a, false, bt[t], (short)0, acc[t], false, false);
        if (kt + 1 < KT) {                    // stage next buffer
            Bsh[cur ^ 1][myidx + 0] = bn0;
            Bsh[cur ^ 1][myidx + 1] = bn1;
            cur ^= 1;
        }
    }

    // epilogue: D layout -> element v is row (v + (lane>=16?8:0)), col (lane&15)
    const int col0 = lane & 15;
    const int mofs = (lane >> 4) * 8;
#pragma unroll
    for (int t = 0; t < 4; ++t) {
        int col = (ntg * 4 + t) * 16 + col0;
        float bv = bias ? bias[col] : 0.f;
#pragma unroll
        for (int v = 0; v < 8; ++v) {
            int r = mt * 16 + mofs + v;
            float val = acc[t][v] + bv;
            if (relu) val = fmaxf(val, 0.f);
            size_t idx = (size_t)r * N + col;
            C[idx] = val;
            if (Cbf) Cbf[idx] = f2bf(val);
        }
    }
}

// ---------------- GRU gate fusion (PyTorch order r,z,n) ----------------
__global__ __launch_bounds__(256) void gru_gates_kernel(const float* __restrict__ gi,
                                                        const float* __restrict__ gh,
                                                        float* __restrict__ hF,
                                                        bf16_t* __restrict__ hB) {
    int i = blockIdx.x * 256 + threadIdx.x;            // over NS*HID
    if (i >= NS * HID) return;
    int b = i / HID, j = i % HID;
    size_t base = (size_t)b * 3 * HID;
    float ir = gi[base + j],          hr = gh[base + j];
    float iz = gi[base + HID + j],    hz = gh[base + HID + j];
    float in_ = gi[base + 2*HID + j], hn = gh[base + 2*HID + j];
    float r  = 1.f / (1.f + __expf(-(ir + hr)));
    float zg = 1.f / (1.f + __expf(-(iz + hz)));
    float n  = tanhf(in_ + r * hn);
    float hp = hF[i];
    float h  = (1.f - zg) * n + zg * hp;
    hF[i] = h;
    hB[i] = f2bf(h);
}

// ---------------- FiLM: outbf = h2 * (1+gamma[j]) + beta[j] ----------------
__global__ __launch_bounds__(256) void film_kernel(const float* __restrict__ h2,
                                                   const float* __restrict__ gamma,
                                                   const float* __restrict__ beta,
                                                   bf16_t* __restrict__ outbf) {
    int i = blockIdx.x * 256 + threadIdx.x;
    if (i >= NS * HID) return;
    int j = i % HID;
    outbf[i] = f2bf(h2[i] * (1.f + gamma[j]) + beta[j]);
}

// ---------------- replicate h0 into 3 layer states ----------------
__global__ __launch_bounds__(256) void repl_h_kernel(const float* __restrict__ h0,
                                                     float* __restrict__ hF,
                                                     bf16_t* __restrict__ hB) {
    int i = blockIdx.x * 256 + threadIdx.x;
    if (i >= NS * HID) return;
    float v = h0[i];
#pragma unroll
    for (int l = 0; l < 3; ++l) {
        hF[(size_t)l * NS * HID + i] = v;
        hB[(size_t)l * NS * HID + i] = f2bf(v);
    }
}

// ---------------- init sequence & tokens ----------------
__global__ __launch_bounds__(256) void init_seq_kernel(float* __restrict__ seq,
                                                       int* __restrict__ toks) {
    int b = blockIdx.x * 256 + threadIdx.x;
    if (b < NS) { seq[(size_t)b * (ML + 1)] = (float)BOS; toks[b] = BOS; }
}

// ---------------- Gumbel-max categorical sampler ----------------
__device__ __forceinline__ unsigned hash3(unsigned a, unsigned b, unsigned c) {
    unsigned x = a * 0x9E3779B9u ^ b * 0x85EBCA6Bu ^ c * 0xC2B2AE35u;
    x ^= x >> 16; x *= 0x7feb352du; x ^= x >> 15; x *= 0x846ca68bu; x ^= x >> 16;
    return x;
}

__global__ __launch_bounds__(VOCAB) void sample_kernel(const float* __restrict__ logits,
                                                       const float* __restrict__ temp,
                                                       const unsigned* __restrict__ key,
                                                       int* __restrict__ toks,
                                                       float* __restrict__ seq, int step) {
    int row = blockIdx.x, v = threadIdx.x;
    float t = fmaxf(1e-8f, temp[0]);
    unsigned seed = key ? key[0] : 42u;
    float l = logits[(size_t)row * VOCAB + v] / t;
    unsigned hu = hash3(seed + (unsigned)step * 0x1000193u, (unsigned)row, (unsigned)v);
    float u = (float)(hu >> 8) * (1.0f / 16777216.0f) + 1e-12f;
    float g = -__logf(-__logf(u));
    float val = l + g;

    __shared__ float sval[VOCAB];
    __shared__ int   sidx[VOCAB];
    sval[v] = val; sidx[v] = v;
    __syncthreads();
    for (int off = VOCAB / 2; off > 0; off >>= 1) {
        if (v < off && sval[v + off] > sval[v]) { sval[v] = sval[v + off]; sidx[v] = sidx[v + off]; }
        __syncthreads();
    }
    if (v == 0) {
        toks[row] = sidx[0];
        seq[(size_t)row * (ML + 1) + step + 1] = (float)sidx[0];
    }
}

// ================= host side =================
static inline size_t alignup(size_t x) { return (x + 255) & ~(size_t)255; }

extern "C" void kernel_launch(void* const* d_in, const int* in_sizes, int n_in,
                              void* d_out, int out_size, void* d_ws, size_t ws_size,
                              hipStream_t stream) {
    (void)in_sizes; (void)out_size; (void)ws_size;
    const float* pe      = (const float*)d_in[0];
    const float* temp    = (const float*)d_in[3];
    const float* embW    = (const float*)d_in[4];
    const float* initW   = (const float*)d_in[5];
    const float* initb   = (const float*)d_in[6];
    const float* Wih[3]  = { (const float*)d_in[7],  (const float*)d_in[8],  (const float*)d_in[9]  };
    const float* Whh[3]  = { (const float*)d_in[10], (const float*)d_in[11], (const float*)d_in[12] };
    const float* bih[3]  = { (const float*)d_in[13], (const float*)d_in[14], (const float*)d_in[15] };
    const float* bhh[3]  = { (const float*)d_in[16], (const float*)d_in[17], (const float*)d_in[18] };
    const float* gammaW  = (const float*)d_in[19];
    const float* gammab  = (const float*)d_in[20];
    const float* betaW   = (const float*)d_in[21];
    const float* betab   = (const float*)d_in[22];
    const float* projW   = (const float*)d_in[23];
    const float* zin     = (const float*)d_in[24];
    const unsigned* key  = (n_in > 25) ? (const unsigned*)d_in[25] : nullptr;
    float* seq = (float*)d_out;

    // ---- workspace bump allocator ----
    char* ws = (char*)d_ws;
    size_t off = 0;
    auto alloc = [&](size_t bytes) { char* p = ws + off; off += alignup(bytes); return p; };

    bf16_t* emb_bf   = (bf16_t*)alloc((size_t)VOCAB * EMB * 2);
    bf16_t* embT_swz = (bf16_t*)alloc((size_t)EMB * VOCAB * 2);     // K=EMB, N=VOCAB
    bf16_t* wih_swz[3];
    wih_swz[0] = (bf16_t*)alloc((size_t)EMB * 3 * HID * 2);
    wih_swz[1] = (bf16_t*)alloc((size_t)HID * 3 * HID * 2);
    wih_swz[2] = (bf16_t*)alloc((size_t)HID * 3 * HID * 2);
    bf16_t* whh_swz[3];
    for (int l = 0; l < 3; ++l) whh_swz[l] = (bf16_t*)alloc((size_t)HID * 3 * HID * 2);
    bf16_t* proj_swz  = (bf16_t*)alloc((size_t)HID * EMB * 2);
    bf16_t* initz_swz = (bf16_t*)alloc((size_t)LAT * HID * 2);
    float*  pe_contrib = (float*)alloc(HID * 4);
    float*  gammaV   = (float*)alloc(HID * 4);
    float*  betaV    = (float*)alloc(HID * 4);
    bf16_t* z_bf     = (bf16_t*)alloc((size_t)NS * LAT * 2);
    float*  h0f      = (float*)alloc((size_t)NS * HID * 4);
    float*  hF       = (float*)alloc((size_t)3 * NS * HID * 4);
    bf16_t* hB       = (bf16_t*)alloc((size_t)3 * NS * HID * 2);
    float*  gi       = (float*)alloc((size_t)NS * 3 * HID * 4);
    float*  gh       = (float*)alloc((size_t)NS * 3 * HID * 4);
    bf16_t* out_bf   = (bf16_t*)alloc((size_t)NS * HID * 2);
    float*  tf       = (float*)alloc((size_t)NS * EMB * 4);
    bf16_t* t_bf     = (bf16_t*)alloc((size_t)NS * EMB * 2);
    float*  logitsf  = (float*)alloc((size_t)NS * VOCAB * 4);
    int*    toks     = (int*)alloc(NS * 4);

    auto grid1 = [](int n) { return dim3((n + 255) / 256); };

    // ---- precompute: converts / swizzles (deterministic, every launch) ----
    cvt_bf16_kernel<<<grid1(VOCAB * EMB), 256, 0, stream>>>(embW, emb_bf, VOCAB * EMB);
    swizzleB_kernel<<<grid1(EMB * VOCAB), 256, 0, stream>>>(embW, embT_swz, EMB, VOCAB, 1);
    swizzleB_kernel<<<grid1(EMB * 3 * HID), 256, 0, stream>>>(Wih[0], wih_swz[0], EMB, 3 * HID, 0);
    for (int l = 1; l < 3; ++l)
        swizzleB_kernel<<<grid1(HID * 3 * HID), 256, 0, stream>>>(Wih[l], wih_swz[l], HID, 3 * HID, 0);
    for (int l = 0; l < 3; ++l)
        swizzleB_kernel<<<grid1(HID * 3 * HID), 256, 0, stream>>>(Whh[l], whh_swz[l], HID, 3 * HID, 0);
    swizzleB_kernel<<<grid1(HID * EMB), 256, 0, stream>>>(projW, proj_swz, HID, EMB, 0);
    swizzleB_kernel<<<grid1(LAT * HID), 256, 0, stream>>>(initW, initz_swz, LAT, HID, 0);

    // FiLM vectors + pe contribution to h0 (pe broadcast over batch -> matvecs)
    matvec_kernel<<<grid1(HID), 256, 0, stream>>>(pe, initW + (size_t)LAT * HID, initb, pe_contrib, PE, HID);
    matvec_kernel<<<grid1(HID), 256, 0, stream>>>(pe, gammaW, gammab, gammaV, PE, HID);
    matvec_kernel<<<grid1(HID), 256, 0, stream>>>(pe, betaW, betab, betaV, PE, HID);

    // h0 = relu(z @ initW[:LAT] + pe_contrib)
    cvt_bf16_kernel<<<grid1(NS * LAT), 256, 0, stream>>>(zin, z_bf, NS * LAT);
    wmma_gemm_lds_kernel<<<dim3(NS / 64, HID / 64), 128, 0, stream>>>(
        z_bf, initz_swz, pe_contrib, h0f, nullptr, nullptr, nullptr, NS, HID, LAT, 1);
    repl_h_kernel<<<grid1(NS * HID), 256, 0, stream>>>(h0f, hF, hB);
    init_seq_kernel<<<grid1(NS), 256, 0, stream>>>(seq, toks);

    // ---- autoregressive loop ----
    const dim3 gGate(NS / 64, (3 * HID) / 64);
    const dim3 gProj(NS / 64, EMB / 64);
    const dim3 gLog(NS / 64, VOCAB / 64);
    for (int step = 0; step < ML; ++step) {
        for (int l = 0; l < 3; ++l) {
            bf16_t* hBl = hB + (size_t)l * NS * HID;
            float*  hFl = hF + (size_t)l * NS * HID;
            if (l == 0) {
                // embedding gather fused into the gi GEMM
                wmma_gemm_lds_kernel<<<gGate, 128, 0, stream>>>(
                    nullptr, wih_swz[0], bih[0], gi, nullptr,
                    toks, emb_bf, NS, 3 * HID, EMB, 0);
            } else {
                wmma_gemm_lds_kernel<<<gGate, 128, 0, stream>>>(
                    hB + (size_t)(l - 1) * NS * HID, wih_swz[l], bih[l], gi, nullptr,
                    nullptr, nullptr, NS, 3 * HID, HID, 0);
            }
            wmma_gemm_lds_kernel<<<gGate, 128, 0, stream>>>(
                hBl, whh_swz[l], bhh[l], gh, nullptr,
                nullptr, nullptr, NS, 3 * HID, HID, 0);
            gru_gates_kernel<<<grid1(NS * HID), 256, 0, stream>>>(gi, gh, hFl, hBl);
        }

        film_kernel<<<grid1(NS * HID), 256, 0, stream>>>(hF + (size_t)2 * NS * HID,
                                                         gammaV, betaV, out_bf);
        wmma_gemm_lds_kernel<<<gProj, 128, 0, stream>>>(out_bf, proj_swz, nullptr,
                                                        tf, t_bf, nullptr, nullptr,
                                                        NS, EMB, HID, 0);
        wmma_gemm_lds_kernel<<<gLog, 128, 0, stream>>>(t_bf, embT_swz, nullptr,
                                                       logitsf, nullptr, nullptr, nullptr,
                                                       NS, VOCAB, EMB, 0);
        sample_kernel<<<NS, VOCAB, 0, stream>>>(logitsf, temp, key, toks, seq, step);
    }
}
